// StyleSwinTransformerBlock_120259084939
// MI455X (gfx1250) — matrix operand order repaired and checked by reading, not compiled, not run on hardware
//
#include <hip/hip_runtime.h>
#include <hip/hip_bf16.h>
#include <math.h>

// ---------------------------------------------------------------- constants
#define B_  8
#define H_  128
#define W_  128
#define C_  256
#define L_  (H_ * W_)          // 16384
#define HEADS_ 8
#define WS_ 8
#define SHIFT_ 4
#define STYLE_ 512
#define MLP_HID_ (4 * C_)      // 1024
#define NW_PER_B_ ((H_/WS_)*(W_/WS_))   // 256
#define NTOK_ (WS_*WS_)        // 64
#define HD_ 32                 // head dim (C/HEADS)

typedef __bf16 bf16;
typedef __attribute__((ext_vector_type(16))) __bf16 v16bf;
typedef __attribute__((ext_vector_type(8)))  float  v8f;
typedef __attribute__((ext_vector_type(4)))  unsigned uint32x4;
typedef __attribute__((ext_vector_type(8)))  int      int32x8;
typedef __attribute__((ext_vector_type(4)))  int      int32x4;

// fragment viewed as two 128-bit chunks -> ds_load_b128 instead of 16x v_mov_b16
union Frag {
    v16bf v;
    uint4 q[2];
};

__device__ __forceinline__ bf16 f2bf(float f) {
    union { float f; unsigned u; } v; v.f = f;
    unsigned r = (v.u + 0x7FFFu + ((v.u >> 16) & 1u)) >> 16;
    union { unsigned short s; bf16 b; } o; o.s = (unsigned short)r;
    return o.b;
}
__device__ __forceinline__ float bf2f(bf16 b) {
    union { unsigned short s; bf16 b; } i; i.b = b;
    union { unsigned u; float f; } o; o.u = ((unsigned)i.s) << 16;
    return o.f;
}

// ---------------------------------------------------------------- TDM 2D load
// DMA a (tile_d1 x tile_d0) tile of 2-byte elements from a row-major tensor
// (row stride = stride_elems) into LDS at lds_addr. D# packing per
// CDNA5 ISA. Tracked by TENSORcnt.
__device__ __forceinline__ void tdm_load_2d_b16(const void* gaddr, unsigned lds_addr,
                                                unsigned tile_d0, unsigned tile_d1,
                                                unsigned tensor_d0, unsigned tensor_d1,
                                                unsigned long long stride_elems) {
    unsigned long long ga = (unsigned long long)(size_t)gaddr;
    uint32x4 g0;
    g0[0] = 1u;                                       // count=1 (valid), user mode
    g0[1] = lds_addr;                                 // LDS byte address
    g0[2] = (unsigned)(ga & 0xFFFFFFFFu);             // global_addr[31:0]
    g0[3] = (unsigned)((ga >> 32) & 0x01FFFFFFu) | (2u << 30);  // addr[56:32], type=2
    int32x8 g1;
    g1[0] = (int)0x10000u;                            // data_size=1 (2 bytes)
    g1[1] = (int)((tensor_d0 & 0xFFFFu) << 16);       // tensor_dim0[15:0]
    g1[2] = (int)((tensor_d0 >> 16) | ((tensor_d1 & 0xFFFFu) << 16));
    g1[3] = (int)((tensor_d1 >> 16) | (tile_d0 << 16));        // tile_dim0
    g1[4] = (int)(tile_d1 & 0xFFFFu);                 // tile_dim1 (tile_dim2 = 0)
    g1[5] = (int)(unsigned)(stride_elems & 0xFFFFFFFFu);       // dim0_stride lo
    g1[6] = (int)((stride_elems >> 32) & 0xFFFFu);    // dim0_stride hi (dim1_stride=0)
    g1[7] = 0;
    int32x4 z4; z4[0] = z4[1] = z4[2] = z4[3] = 0;
#if defined(__clang_major__) && (__clang_major__ >= 23)
    int32x8 z8; z8[0]=z8[1]=z8[2]=z8[3]=z8[4]=z8[5]=z8[6]=z8[7]=0;
    __builtin_amdgcn_tensor_load_to_lds(g0, g1, z4, z4, z8, 0);
#else
    __builtin_amdgcn_tensor_load_to_lds(g0, g1, z4, z4, 0);
#endif
}

// ---------------------------------------------------------------- fp32 -> bf16
__global__ __launch_bounds__(256)
void cvt_f32_bf16(const float* __restrict__ src, bf16* __restrict__ dst, int n) {
    int i = blockIdx.x * 256 + threadIdx.x;
    if (i < n) dst[i] = f2bf(src[i]);
}

// ---------------------------------------------------------------- style proj
__global__ __launch_bounds__(256)
void style_proj(const float* __restrict__ style, const float* __restrict__ w,
                const float* __restrict__ bias, float* __restrict__ gamma,
                float* __restrict__ beta) {
    int b = blockIdx.x;
    __shared__ float s[STYLE_];
    s[threadIdx.x]       = style[b * STYLE_ + threadIdx.x];
    s[threadIdx.x + 256] = style[b * STYLE_ + 256 + threadIdx.x];
    __syncthreads();
    const float scale = 0.044194173824159216f;  // 1/sqrt(512)
    for (int o = threadIdx.x; o < 2 * C_; o += 256) {
        const float* wr = w + (size_t)o * STYLE_;
        float acc = 0.f;
        #pragma unroll 4
        for (int k = 0; k < STYLE_; ++k) acc += s[k] * wr[k];
        acc = acc * scale + bias[o];
        if (o < C_) gamma[b * C_ + o] = acc;
        else        beta[b * C_ + (o - C_)] = acc;
    }
}

// ---------------------------------------------------------------- mean / rvar
// stage 1: coalesced (thread = channel) partial sums over 256-token chunks
__global__ __launch_bounds__(256)
void mean_var_part(const float* __restrict__ x, float* __restrict__ ps,
                   float* __restrict__ ps2) {
    const int b = blockIdx.y, chunk = blockIdx.x;   // 64 chunks x 256 tokens
    const int c = threadIdx.x;
    const float* p = x + ((size_t)b * L_ + (size_t)chunk * 256) * C_ + c;
    float s = 0.f, s2 = 0.f;
    #pragma unroll 4
    for (int l = 0; l < 256; ++l) {
        float v = p[(size_t)l * C_];
        s += v; s2 += v * v;
    }
    size_t o = ((size_t)b * 64 + chunk) * C_ + c;
    ps[o] = s; ps2[o] = s2;
}
// stage 2: fold 64 partials per (b,c)
__global__ __launch_bounds__(256)
void mean_var_final(const float* __restrict__ ps, const float* __restrict__ ps2,
                    float* __restrict__ mu, float* __restrict__ rv) {
    int bc = blockIdx.x * 256 + threadIdx.x;        // 0..2047
    int b = bc >> 8, c = bc & (C_ - 1);
    float s = 0.f, s2 = 0.f;
    #pragma unroll 8
    for (int k = 0; k < 64; ++k) {
        size_t o = ((size_t)b * 64 + k) * C_ + c;
        s += ps[o]; s2 += ps2[o];
    }
    float m = s * (1.f / L_);
    float v = s2 * (1.f / L_) - m * m;
    mu[bc] = m;
    rv[bc] = rsqrtf(v + 1e-5f);
}

// ---------------------------------------------------------------- AdaIN -> bf16
__global__ __launch_bounds__(256)
void adain_norm(const float* __restrict__ x, const float* __restrict__ gamma,
                const float* __restrict__ beta, const float* __restrict__ mu,
                const float* __restrict__ rv, bf16* __restrict__ out) {
    int idx = blockIdx.x * 256 + threadIdx.x;       // < B*L*C = 2^25
    int c  = idx & (C_ - 1);
    int b  = idx >> 22;                             // / (L*C)
    int bc = b * C_ + c;
    out[idx] = f2bf(gamma[bc] * (x[idx] - mu[bc]) * rv[bc] + beta[bc]);
}

// ---------------------------------------------------------------- WMMA GEMM
// C(MxN) = A(MxK,bf16,row) @ W(NxK,bf16,row)^T  (+ epilogue)
// block tile 128x128, 8 waves (2 M x 4 N), wave tile 64x32 = 4x2 wmma tiles.
// Both LDS slabs are staged by the Tensor Data Mover (one 2D descriptor each,
// issued by wave 0); waves only do ds_load_b128 + v_wmma.
#define BM 128
#define BN 128
#define BK 32
// EPI: 0 = +bias -> bf16 ; 1 = +bias +resid(f32) -> f32 ; 2 = +bias, GELU -> bf16
template <int EPI>
__global__ __launch_bounds__(256)
void gemm_bf16(const bf16* __restrict__ A, const bf16* __restrict__ Wt,
               const float* __restrict__ bias, const float* __restrict__ resid,
               float* __restrict__ outF, bf16* __restrict__ outB,
               int M, int N, int K) {
    __shared__ bf16 sA[BM * BK];
    __shared__ bf16 sB[BN * BK];

    const int tid  = threadIdx.x;
    const int lane = tid & 31;
    const int wave = tid >> 5;
    const int m0 = blockIdx.y * BM;
    const int n0 = blockIdx.x * BN;
    const int waveM = wave & 1;     // 0..1 -> 64-row strip
    const int waveN = wave >> 1;    // 0..3 -> 32-col strip

    v8f acc[4][2];
    #pragma unroll
    for (int i = 0; i < 4; ++i)
        #pragma unroll
        for (int j = 0; j < 2; ++j)
            #pragma unroll
            for (int e = 0; e < 8; ++e) acc[i][j][e] = 0.f;

    const int h8   = (lane >> 4) * 8;
    const int h16  = (lane >> 4) * 16;
    const int rl   = lane & 15;
    const unsigned ldsA = (unsigned)(size_t)&sA[0];
    const unsigned ldsB = (unsigned)(size_t)&sB[0];

    for (int k0 = 0; k0 < K; k0 += BK) {
        // stage A/B slabs into LDS via TDM (async tensor DMA, TENSORcnt)
        if (wave == 0) {
            tdm_load_2d_b16(A  + (size_t)m0 * K + k0, ldsA, BK, BM,
                            (unsigned)K, (unsigned)M, (unsigned long long)K);
            tdm_load_2d_b16(Wt + (size_t)n0 * K + k0, ldsB, BK, BN,
                            (unsigned)K, (unsigned)N, (unsigned long long)K);
            __builtin_amdgcn_s_wait_tensorcnt(0);
        }
        __syncthreads();

        // B fragments: two contiguous 8-elem groups at col*BK + h8 and +16 -> two ds_load_b128
        Frag bfrag[2];
        #pragma unroll
        for (int j = 0; j < 2; ++j) {
            const bf16* p = &sB[(waveN * 32 + j * 16 + rl) * BK + h8];
            bfrag[j].q[0] = *(const uint4*)p;
            bfrag[j].q[1] = *(const uint4*)(p + 16);
        }
        #pragma unroll
        for (int i = 0; i < 4; ++i) {
            // A fragment: two contiguous 8-elem groups -> two ds_load_b128
            const bf16* p = &sA[(waveM * 64 + i * 16 + rl) * BK + h8];
            Frag af;
            af.q[0] = *(const uint4*)p;
            af.q[1] = *(const uint4*)(p + 16);
            #pragma unroll
            for (int j = 0; j < 2; ++j)
                acc[i][j] = __builtin_amdgcn_wmma_f32_16x16x32_bf16(
                    false, af.v, false, bfrag[j].v, (short)0, acc[i][j], false, false);
        }
        __syncthreads();
    }

    // epilogue: C layout -> VGPR r, lanes 0-15: (M=r, N=lane), lanes 16-31: (M=8+r)
    const int half = lane >> 4;
    #pragma unroll
    for (int i = 0; i < 4; ++i) {
        #pragma unroll
        for (int j = 0; j < 2; ++j) {
            int col = n0 + waveN * 32 + j * 16 + rl;
            float bv = bias[col];
            #pragma unroll
            for (int r = 0; r < 8; ++r) {
                int row = m0 + waveM * 64 + i * 16 + half * 8 + r;
                size_t o = (size_t)row * N + col;
                float v = acc[i][j][r] + bv;
                if (EPI == 0) {
                    outB[o] = f2bf(v);
                } else if (EPI == 1) {
                    outF[o] = v + resid[o];
                } else {  // GELU (exact)
                    outB[o] = f2bf(0.5f * v * (1.f + erff(v * 0.70710678118654752f)));
                }
            }
        }
    }
}

// ---------------------------------------------------------------- window attn
// grid: (B*NW, HEADS/2, 2 halves); block: 64 threads (2 waves)
__global__ __launch_bounds__(64)
void win_attn(const bf16* __restrict__ qkv, const float* __restrict__ rpb1,
              const float* __restrict__ rpb2, bf16* __restrict__ xa) {
    __shared__ bf16  Qs[NTOK_ * HD_];    // 64x32 token-major (scaled)
    __shared__ bf16  Ks[NTOK_ * HD_];    // 64x32 token-major
    __shared__ bf16  Vt[HD_ * NTOK_];    // 32x64 dim-major (for P@V B-frags)
    __shared__ float Ss[NTOK_ * NTOK_];  // 64x64 logits
    __shared__ bf16  Pb[NTOK_ * NTOK_];  // 64x64 probs

    const int tid  = threadIdx.x;
    const int lane = tid & 31;
    const int wave = tid >> 5;
    const int wflat = blockIdx.x;        // b * 256 + win
    const int head  = blockIdx.y;        // 0..3 (within half)
    const int hs    = blockIdx.z;        // 0 = plain, 1 = shifted
    const int b   = wflat >> 8;
    const int win = wflat & 255;
    const int wi  = win >> 4, wj = win & 15;
    const int chq = hs * (C_ / 2) + head * HD_;

    // ---- load Q (scaled), K (row-major), V^T for this window/head
    {
        const int t = tid;              // token 0..63
        const int i = t >> 3, j = t & 7;
        int h = wi * WS_ + i, w = wj * WS_ + j;
        if (hs) { h = (h + SHIFT_) & (H_ - 1); w = (w + SHIFT_) & (W_ - 1); }
        size_t rowoff = ((size_t)b * L_ + (size_t)h * W_ + w) * (3 * C_);
        const bf16* qp = qkv + rowoff + chq;
        const bf16* kp = qkv + rowoff + C_ + chq;
        const bf16* vp = qkv + rowoff + 2 * C_ + chq;
        const float scale = 0.17677669529663687f;   // 1/sqrt(32)
        // K: straight 64B vector copy (global b128 -> ds b128)
        #pragma unroll
        for (int u = 0; u < 4; ++u)
            ((uint4*)&Ks[t * HD_])[u] = ((const uint4*)kp)[u];
        #pragma unroll
        for (int c = 0; c < HD_; ++c) {
            Qs[t * HD_ + c]   = f2bf(bf2f(qp[c]) * scale);
            Vt[c * NTOK_ + t] = vp[c];
        }
    }
    __syncthreads();

    const int h8   = (lane >> 4) * 8;
    const int h16  = (lane >> 4) * 16;
    const int rl   = lane & 15;
    const int half = lane >> 4;

    // ---- S = Q @ K^T  (4x4 tiles of 16x16, K=32: one wmma each)
    #pragma unroll
    for (int mi = 0; mi < 2; ++mi) {
        int mt = wave * 2 + mi;
        const bf16* ap = &Qs[(mt * 16 + rl) * HD_ + h8];
        Frag af;
        af.q[0] = *(const uint4*)ap;
        af.q[1] = *(const uint4*)(ap + 16);
        #pragma unroll
        for (int nt = 0; nt < 4; ++nt) {
            const bf16* bp = &Ks[(nt * 16 + rl) * HD_ + h8];
            Frag bfr;
            bfr.q[0] = *(const uint4*)bp;
            bfr.q[1] = *(const uint4*)(bp + 16);
            v8f c;
            #pragma unroll
            for (int e = 0; e < 8; ++e) c[e] = 0.f;
            c = __builtin_amdgcn_wmma_f32_16x16x32_bf16(
                false, af.v, false, bfr.v, (short)0, c, false, false);
            #pragma unroll
            for (int r = 0; r < 8; ++r)
                Ss[(mt * 16 + half * 8 + r) * NTOK_ + nt * 16 + rl] = c[r];
        }
    }
    __syncthreads();

    // ---- softmax over keys; one row per thread; bias + shift mask inline
    {
        const int n = tid;
        const int qi = n >> 3, qj = n & 7;
        const float* rpb = hs ? rpb2 : rpb1;
        int regn = 0;
        if (hs) {
            int r = wi * WS_ + qi, cc = wj * WS_ + qj;
            int rr = (r  < H_ - WS_) ? 0 : ((r  < H_ - SHIFT_) ? 1 : 2);
            int rc = (cc < W_ - WS_) ? 0 : ((cc < W_ - SHIFT_) ? 1 : 2);
            regn = rr * 3 + rc;
        }
        float logit[NTOK_];
        float mx = -1e30f;
        for (int m = 0; m < NTOK_; ++m) {
            int ki = m >> 3, kj = m & 7;
            int rel = ((qi - ki) + 7) * 15 + ((qj - kj) + 7);
            float lv = Ss[n * NTOK_ + m] + rpb[rel * 4 + head];
            if (hs) {
                int r = wi * WS_ + ki, cc = wj * WS_ + kj;
                int rr = (r  < H_ - WS_) ? 0 : ((r  < H_ - SHIFT_) ? 1 : 2);
                int rc = (cc < W_ - WS_) ? 0 : ((cc < W_ - SHIFT_) ? 1 : 2);
                if (rr * 3 + rc != regn) lv -= 100.f;
            }
            logit[m] = lv;
            mx = fmaxf(mx, lv);
        }
        float s = 0.f;
        for (int m = 0; m < NTOK_; ++m) { float e = __expf(logit[m] - mx); logit[m] = e; s += e; }
        float inv = 1.f / s;
        for (int m = 0; m < NTOK_; ++m) Pb[n * NTOK_ + m] = f2bf(logit[m] * inv);
    }
    __syncthreads();

    // ---- O = P @ V  (4x2 tiles, K=64 -> two wmma); write rolled-back layout
    #pragma unroll
    for (int mi = 0; mi < 2; ++mi) {
        int mt = wave * 2 + mi;
        #pragma unroll
        for (int nt = 0; nt < 2; ++nt) {
            v8f c;
            #pragma unroll
            for (int e = 0; e < 8; ++e) c[e] = 0.f;
            #pragma unroll
            for (int kt = 0; kt < 2; ++kt) {
                int kb = kt * 32;
                const bf16* ap = &Pb[(mt * 16 + rl) * NTOK_ + kb + h8];
                Frag af;
                af.q[0] = *(const uint4*)ap;
                af.q[1] = *(const uint4*)(ap + 16);
                const bf16* bp = &Vt[(nt * 16 + rl) * NTOK_ + kb + h8];
                Frag bfr;
                bfr.q[0] = *(const uint4*)bp;
                bfr.q[1] = *(const uint4*)(bp + 16);
                c = __builtin_amdgcn_wmma_f32_16x16x32_bf16(
                    false, af.v, false, bfr.v, (short)0, c, false, false);
            }
            #pragma unroll
            for (int r = 0; r < 8; ++r) {
                int n = mt * 16 + half * 8 + r;      // token in window
                int d = nt * 16 + rl;
                int i = n >> 3, j = n & 7;
                int h = wi * WS_ + i, w = wj * WS_ + j;
                if (hs) { h = (h + SHIFT_) & (H_ - 1); w = (w + SHIFT_) & (W_ - 1); }
                size_t o = ((size_t)b * L_ + (size_t)h * W_ + w) * C_ + chq + d;
                xa[o] = f2bf(c[r]);
            }
        }
    }
}

// ---------------------------------------------------------------- launcher
extern "C" void kernel_launch(void* const* d_in, const int* in_sizes, int n_in,
                              void* d_out, int out_size, void* d_ws, size_t ws_size,
                              hipStream_t stream) {
    (void)in_sizes; (void)n_in; (void)out_size; (void)ws_size;
    const float* x       = (const float*)d_in[0];
    const float* style   = (const float*)d_in[1];
    const float* wst1    = (const float*)d_in[2];
    const float* bst1    = (const float*)d_in[3];
    const float* qkv_w   = (const float*)d_in[4];
    const float* qkv_b   = (const float*)d_in[5];
    const float* rpb1    = (const float*)d_in[6];
    const float* rpb2    = (const float*)d_in[7];
    const float* proj_w  = (const float*)d_in[8];
    const float* proj_b  = (const float*)d_in[9];
    const float* wst2    = (const float*)d_in[10];
    const float* bst2    = (const float*)d_in[11];
    const float* fc1_w   = (const float*)d_in[12];
    const float* fc1_b   = (const float*)d_in[13];
    const float* fc2_w   = (const float*)d_in[14];
    const float* fc2_b   = (const float*)d_in[15];
    float* out = (float*)d_out;

    // ---- workspace layout (256B aligned)
    char* ws = (char*)d_ws;
    size_t off = 0;
    auto alloc = [&](size_t bytes) { void* p = ws + off; off += (bytes + 255) & ~(size_t)255; return p; };
    bf16*  wqkv_bf  = (bf16*)alloc((size_t)(3*C_)*C_*2);
    bf16*  wproj_bf = (bf16*)alloc((size_t)C_*C_*2);
    bf16*  wfc1_bf  = (bf16*)alloc((size_t)MLP_HID_*C_*2);
    bf16*  wfc2_bf  = (bf16*)alloc((size_t)C_*MLP_HID_*2);
    float* gamma1   = (float*)alloc((size_t)B_*C_*4);
    float* beta1    = (float*)alloc((size_t)B_*C_*4);
    float* gamma2   = (float*)alloc((size_t)B_*C_*4);
    float* beta2    = (float*)alloc((size_t)B_*C_*4);
    float* mu       = (float*)alloc((size_t)B_*C_*4);
    float* rv       = (float*)alloc((size_t)B_*C_*4);
    float* part1    = (float*)alloc((size_t)B_*64*C_*4);
    float* part2    = (float*)alloc((size_t)B_*64*C_*4);
    bf16*  xn_bf    = (bf16*)alloc((size_t)B_*L_*C_*2);        // reused for yn
    bf16*  qkv_bf   = (bf16*)alloc((size_t)B_*L_*3*C_*2);
    bf16*  xa_bf    = (bf16*)alloc((size_t)B_*L_*C_*2);
    float* y        = (float*)alloc((size_t)B_*L_*C_*4);
    bf16*  h_bf     = (bf16*)alloc((size_t)B_*L_*MLP_HID_*2);

    const int M = B_ * L_;                 // 131072
    const int total = B_ * L_ * C_;        // 33554432

    // 1) weights -> bf16
    cvt_f32_bf16<<<(3*C_*C_ + 255)/256, 256, 0, stream>>>(qkv_w,  wqkv_bf,  3*C_*C_);
    cvt_f32_bf16<<<(C_*C_   + 255)/256, 256, 0, stream>>>(proj_w, wproj_bf, C_*C_);
    cvt_f32_bf16<<<(MLP_HID_*C_ + 255)/256, 256, 0, stream>>>(fc1_w, wfc1_bf, MLP_HID_*C_);
    cvt_f32_bf16<<<(C_*MLP_HID_ + 255)/256, 256, 0, stream>>>(fc2_w, wfc2_bf, C_*MLP_HID_);

    // 2) style -> gamma/beta (both layers)
    style_proj<<<B_, 256, 0, stream>>>(style, wst1, bst1, gamma1, beta1);
    style_proj<<<B_, 256, 0, stream>>>(style, wst2, bst2, gamma2, beta2);

    // 3) AdaIN #1 -> xn (bf16)
    mean_var_part<<<dim3(64, B_), 256, 0, stream>>>(x, part1, part2);
    mean_var_final<<<(B_*C_)/256, 256, 0, stream>>>(part1, part2, mu, rv);
    adain_norm<<<total/256, 256, 0, stream>>>(x, gamma1, beta1, mu, rv, xn_bf);

    // 4) QKV = xn @ qkv_w^T + b   (M x 768)
    gemm_bf16<0><<<dim3(3*C_/BN, M/BM), 256, 0, stream>>>(
        xn_bf, wqkv_bf, qkv_b, nullptr, nullptr, qkv_bf, M, 3*C_, C_);

    // 5) dual-window attention -> xa (bf16, already window-reversed/rolled/concat)
    win_attn<<<dim3(B_*NW_PER_B_, HEADS_/2, 2), 64, 0, stream>>>(qkv_bf, rpb1, rpb2, xa_bf);

    // 6) y = shortcut + xa @ proj_w^T + b    (f32)
    gemm_bf16<1><<<dim3(C_/BN, M/BM), 256, 0, stream>>>(
        xa_bf, wproj_bf, proj_b, x, y, nullptr, M, C_, C_);

    // 7) AdaIN #2 -> yn (bf16, reuse xn buffer)
    mean_var_part<<<dim3(64, B_), 256, 0, stream>>>(y, part1, part2);
    mean_var_final<<<(B_*C_)/256, 256, 0, stream>>>(part1, part2, mu, rv);
    adain_norm<<<total/256, 256, 0, stream>>>(y, gamma2, beta2, mu, rv, xn_bf);

    // 8) h = gelu(yn @ fc1^T + b)   (M x 1024, bf16)
    gemm_bf16<2><<<dim3(MLP_HID_/BN, M/BM), 256, 0, stream>>>(
        xn_bf, wfc1_bf, fc1_b, nullptr, nullptr, h_bf, M, MLP_HID_, C_);

    // 9) out = y + h @ fc2^T + b   (f32)
    gemm_bf16<1><<<dim3(C_/BN, M/BM), 256, 0, stream>>>(
        h_bf, wfc2_bf, fc2_b, y, out, nullptr, M, C_, MLP_HID_);
}
